// ASPOC_GSECAM_NetHead_50646254355138
// MI455X (gfx1250) — matrix operand order repaired and checked by reading, not compiled, not run on hardware
//
#include <hip/hip_runtime.h>
#include <hip/hip_bf16.h>
#include <cstdint>
#include <cstddef>

// ---------------------------------------------------------------------------
// CDNA5 / gfx1250 bf16 WMMA implementation of the ASPOC+GSE-CAM head.
// Heavy ops: NT GEMMs on v_wmma_f32_16x16x32_bf16 (wave32).
// Dilated 3x3 convs: cooperative implicit GEMM with async global->LDS staging.
// ---------------------------------------------------------------------------

typedef __bf16 bf16;
typedef __attribute__((ext_vector_type(16))) __bf16 v16bf;
typedef __attribute__((ext_vector_type(8)))  __bf16 v8bf;
typedef __attribute__((ext_vector_type(8)))  float  v8f;
typedef __attribute__((ext_vector_type(4)))  int    v4i;

#ifndef __has_builtin
#define __has_builtin(x) 0
#endif

#if __has_builtin(__builtin_amdgcn_global_load_async_to_lds_b128)
#define HAVE_ASYNC_LDS 1
#else
#define HAVE_ASYNC_LDS 0
#endif

__device__ __forceinline__ v8f zero8() {
  v8f z;
#pragma unroll
  for (int i = 0; i < 8; ++i) z[i] = 0.f;
  return z;
}

__device__ __forceinline__ v8f wmma_bf16(v16bf a, v16bf b, v8f c) {
  // D = A(16x32) * B(32x16) + C, f32 accumulate
  return __builtin_amdgcn_wmma_f32_16x16x32_bf16(
      /*neg_a=*/false, a, /*neg_b=*/false, b,
      /*c_mod=*/(short)0, c, /*reuse_a=*/false, /*reuse_b=*/false);
}

// A fragment: row m, 16 elems at k0+16*half (matches B fragment K order)
__device__ __forceinline__ v16bf load_a_frag(const bf16* __restrict__ A, int lda,
                                             long long m, int k0, int hl) {
  union { v16bf v; v8bf h[2]; } u;
  const bf16* p = A + m * (long long)lda + k0 + 16 * hl;
  u.h[0] = *(const v8bf*)p;
  u.h[1] = *(const v8bf*)(p + 8);
  return u.v;
}

// B fragment: column n (stored K-contiguous), 16 elems at k0+16*half
__device__ __forceinline__ v16bf load_b_frag(const bf16* __restrict__ Bt, long long ldb,
                                             long long n, int k0, int hl) {
  return *(const v16bf*)(Bt + n * ldb + k0 + 16 * hl);
}

// 16-byte global->LDS copy: async DMA on gfx1250 when available.
// Builtin signature (probe-verified): (AS1 v4i*, AS3 v4i*, imm offset, imm cpol)
__device__ __forceinline__ void copy16_g2l(const void* gp, void* lp) {
#if HAVE_ASYNC_LDS
  __builtin_amdgcn_global_load_async_to_lds_b128(
      (__attribute__((address_space(1))) v4i*)const_cast<void*>(gp),
      (__attribute__((address_space(3))) v4i*)lp, 0, 0);
#else
  *(v8bf*)lp = *(const v8bf*)gp;
#endif
}

__device__ __forceinline__ void wait_async0() {
#if HAVE_ASYNC_LDS
#if __has_builtin(__builtin_amdgcn_s_wait_asynccnt)
  __builtin_amdgcn_s_wait_asynccnt(0);
#else
  asm volatile("s_wait_asynccnt 0" ::: "memory");
#endif
#endif
}

// ---------------------------------------------------------------------------
// Generic batched NT GEMM: out[b][m][n] = epi( alpha * sum_k A[m][k]*Bt[n][k] )
// epi: (+bias[m]) -> (*scale[m]+shift[m]) -> relu.  N % 64 == 0, K % 32 == 0.
// Per-wave tile: 16(M) x 64(N).  Simple loop body: the scheduler pipelines
// the B-tile loads against the WMMAs better than manual register ping-pong.
// ---------------------------------------------------------------------------
__global__ void gemm_nt(const bf16* __restrict__ A, const bf16* __restrict__ Bt,
                        float* __restrict__ outF, bf16* __restrict__ oMN,
                        bf16* __restrict__ oNM,
                        int Bn, int M, int N, int K,
                        int lda, int ldb, int ldoF, int ldoB,
                        long long sA, long long sB, long long sOF, long long sOB,
                        const float* __restrict__ bias,
                        const float* __restrict__ scale,
                        const float* __restrict__ shift,
                        int relu, float alpha) {
  const int wave = blockIdx.x * (blockDim.x >> 5) + (threadIdx.x >> 5);
  const int lane = threadIdx.x & 31;
  const int mt_count = (M + 15) >> 4;
  const int nt_count = N >> 6;
  const int per_b = mt_count * nt_count;
  const int b = wave / per_b;
  if (b >= Bn) return;
  const int rem = wave % per_b;
  const int mt = rem / nt_count, nt = rem % nt_count;
  const int l16 = lane & 15, hl = lane >> 4;
  const int m0 = mt * 16, n0 = nt * 64;

  const bf16* Ab = A + (long long)b * sA;
  const bf16* Bb = Bt + (long long)b * sB;

  v8f acc[4];
#pragma unroll
  for (int t = 0; t < 4; ++t) acc[t] = zero8();

  long long mrow = m0 + l16;
  if (mrow >= M) mrow = M - 1;  // clamp (store is guarded)

  for (int k0 = 0; k0 < K; k0 += 32) {
    v16bf a = load_a_frag(Ab, lda, mrow, k0, hl);
    if (k0 + 32 < K)
      __builtin_prefetch(Ab + mrow * (long long)lda + k0 + 32, 0, 3);
#pragma unroll
    for (int t = 0; t < 4; ++t) {
      v16bf bfrag = load_b_frag(Bb, ldb, (long long)(n0 + t * 16 + l16), k0, hl);
      acc[t] = wmma_bf16(a, bfrag, acc[t]);
    }
  }

#pragma unroll
  for (int t = 0; t < 4; ++t) {
    const int n = n0 + t * 16 + l16;
#pragma unroll
    for (int r = 0; r < 8; ++r) {
      const int m = m0 + 8 * hl + r;  // C/D layout: lanes16-31 hold M=r+8
      if (m >= M) continue;
      float v = acc[t][r] * alpha;
      if (bias) v += bias[m];
      if (scale) v = v * scale[m] + shift[m];
      if (relu) v = v > 0.f ? v : 0.f;
      if (outF) outF[(long long)b * sOF + (long long)m * ldoF + n] = v;
      if (oMN) oMN[(long long)b * sOB + (long long)m * ldoB + n] = (bf16)v;
      if (oNM) oNM[(long long)b * sOB + (long long)n * ldoB + m] = (bf16)v;
    }
  }
}

// ---------------------------------------------------------------------------
// Dilated 3x3 conv as 9 shifted accumulated GEMMs — cooperative version.
// One 256-thread block (8 waves) = one image row (64 cols) x 128 out-channels.
// Per (tap,k0) step the 64x32 bf16 activation slab (4KB) is staged into LDS
// via async global->LDS DMA (double buffered) and shared by all 8 waves.
// Wt: [9][256][512] bf16.  Xp: [B][4096][512] bf16 (n-major).
// out = relu(scale*acc + shift), f32 NCHW slice (ldo = 4096).
// ---------------------------------------------------------------------------
__global__ void __launch_bounds__(256)
conv3x3_wmma_lds(const bf16* __restrict__ Wt, const bf16* __restrict__ Xp,
                 float* __restrict__ outF, long long sOF,
                 const float* __restrict__ scale,
                 const float* __restrict__ shift,
                 int dil, int Bn) {
  __shared__ __attribute__((aligned(64))) bf16 sB[2][64][32];  // 8KB double buffer

  const int tid = threadIdx.x;
  const int wav = tid >> 5, lane = tid & 31;
  const int l16 = lane & 15, hl = lane >> 4;

  int blk = blockIdx.x;
  const int row = blk & 63;  blk >>= 6;   // image row h
  const int mhalf = blk & 1; blk >>= 1;   // 0: channels 0..127, 1: 128..255
  const int b = blk;
  if (b >= Bn) return;

  const int m0 = mhalf * 128 + wav * 16;
  const bf16* Xb = Xp + (long long)b * 4096 * 512;
  const int lcol = tid >> 2;          // column (0..63) staged by this thread
  const int lchunk = (tid & 3) * 8;   // 16B sub-chunk within the 32-K slab

  v8f acc[4];
#pragma unroll
  for (int t = 0; t < 4; ++t) acc[t] = zero8();
  const long long mrow = m0 + l16;

  auto stage = [&](int s, int buf) {
    const int tap = s >> 4;            // 9 taps
    const int k0 = (s & 15) << 5;      // 16 k-steps of 32
    const int dy = (tap / 3 - 1) * dil;
    const int dx = (tap % 3 - 1) * dil;
    const int h2 = row + dy;
    const int w2 = lcol + dx;
    bf16* lp = &sB[buf][lcol][lchunk];
    if (h2 >= 0 && h2 < 64 && w2 >= 0 && w2 < 64) {
      const bf16* gp = Xb + (long long)(h2 * 64 + w2) * 512 + k0 + lchunk;
      copy16_g2l(gp, lp);
    } else {
      union { v8bf v; unsigned u[4]; } z;
      z.u[0] = z.u[1] = z.u[2] = z.u[3] = 0u;
      *(v8bf*)lp = z.v;  // zero-pad border taps
    }
  };

  stage(0, 0);
  wait_async0();
  __syncthreads();

  int buf = 0;
  for (int s = 0; s < 9 * 16; ++s) {
    if (s + 1 < 9 * 16) stage(s + 1, buf ^ 1);  // prefetch next slab

    const int tap = s >> 4;
    const int k0 = (s & 15) << 5;
    const bf16* Atap = Wt + (long long)tap * 256 * 512;
    const v16bf a = load_a_frag(Atap, 512, mrow, k0, hl);
#pragma unroll
    for (int t = 0; t < 4; ++t) {
      const v16bf bfrag = *(const v16bf*)&sB[buf][t * 16 + l16][16 * hl];
      acc[t] = wmma_bf16(a, bfrag, acc[t]);
    }

    wait_async0();     // next slab landed in LDS
    __syncthreads();   // everyone done reading current slab
    buf ^= 1;
  }

#pragma unroll
  for (int t = 0; t < 4; ++t) {
    const int n = row * 64 + t * 16 + l16;
#pragma unroll
    for (int r = 0; r < 8; ++r) {
      const int m = m0 + 8 * hl + r;
      float v = acc[t][r] * scale[m] + shift[m];
      v = v > 0.f ? v : 0.f;
      outF[(long long)b * sOF + (long long)m * 4096 + n] = v;
    }
  }
}

// ---------------------------------------------------------------------------
// LDS-tiled transpose+convert: f32 [B][C][N] -> bf16 [B][N][C] and bf16 [B][C][N]
// ---------------------------------------------------------------------------
__global__ void transpose_cvt(const float* __restrict__ in, bf16* __restrict__ outT,
                              bf16* __restrict__ outC, int C, int N) {
  __shared__ float tile[32][33];
  const int b = blockIdx.z;
  const int c0 = blockIdx.y * 32, n0 = blockIdx.x * 32;
  const int tx = threadIdx.x, ty = threadIdx.y;
  const float v = in[((long long)b * C + c0 + ty) * N + n0 + tx];
  tile[ty][tx] = v;
  if (outC) outC[((long long)b * C + c0 + ty) * N + n0 + tx] = (bf16)v;
  __syncthreads();
  if (outT) outT[((long long)b * N + n0 + ty) * C + c0 + tx] = (bf16)tile[tx][ty];
}

__global__ void cvt_plain(const float* __restrict__ in, bf16* __restrict__ out, int n) {
  const int i = blockIdx.x * blockDim.x + threadIdx.x;
  if (i < n) out[i] = (bf16)in[i];
}

// [O][C][3][3] f32 -> [9][O][C] bf16
__global__ void cvt_w3x3(const float* __restrict__ in, bf16* __restrict__ out,
                         int O, int C) {
  const int i = blockIdx.x * blockDim.x + threadIdx.x;
  const int tot = 9 * O * C;
  if (i >= tot) return;
  const int tap = i / (O * C);
  const int rem = i - tap * O * C;
  const int o = rem / C, c = rem % C;
  out[i] = (bf16)in[((long long)o * C + c) * 9 + tap];
}

// ctx_in f32 [B][256][64][64] -> 2x2 maxpool -> bf16 [B][1024][256] (n-major)
__global__ void maxpool_t(const float* __restrict__ in, bf16* __restrict__ outT) {
  const int i = blockIdx.x * blockDim.x + threadIdx.x;
  if (i >= 2 * 1024 * 256) return;
  const int c = i & 255, n2 = (i >> 8) & 1023, b = i >> 18;
  const int h2 = n2 >> 5, w2 = n2 & 31;
  const long long base = ((long long)(b * 256 + c)) * 4096 + (h2 * 2) * 64 + w2 * 2;
  float m = in[base];
  m = fmaxf(m, in[base + 1]);
  m = fmaxf(m, in[base + 64]);
  m = fmaxf(m, in[base + 65]);
  outT[i] = (bf16)m;
}

// rows of f32 -> softmax(alpha*row) -> bf16 (alpha=-1 gives the CAM "max-e" softmax)
__global__ void softmax_rows(const float* __restrict__ in, bf16* __restrict__ out,
                             int cols, float alpha) {
  const long long row = blockIdx.x;
  const float* ip = in + row * cols;
  bf16* op = out + row * cols;
  __shared__ float red[256];
  const int tid = threadIdx.x;
  float mx = -3.4e38f;
  for (int c = tid; c < cols; c += 256) {
    const float v = ip[c] * alpha;
    mx = v > mx ? v : mx;
  }
  red[tid] = mx;
  __syncthreads();
  for (int s = 128; s > 0; s >>= 1) {
    if (tid < s) red[tid] = fmaxf(red[tid], red[tid + s]);
    __syncthreads();
  }
  mx = red[0];
  __syncthreads();
  float sum = 0.f;
  for (int c = tid; c < cols; c += 256) sum += __expf(ip[c] * alpha - mx);
  red[tid] = sum;
  __syncthreads();
  for (int s = 128; s > 0; s >>= 1) {
    if (tid < s) red[tid] += red[tid + s];
    __syncthreads();
  }
  const float inv = 1.f / red[0];
  for (int c = tid; c < cols; c += 256)
    op[c] = (bf16)(__expf(ip[c] * alpha - mx) * inv);
}

// global average pool: [rows][N] f32 -> [rows] f32 mean
__global__ void gap_mean(const float* __restrict__ in, float* __restrict__ out, int N) {
  const long long rc = blockIdx.x;
  const float* ip = in + rc * N;
  __shared__ float red[256];
  const int tid = threadIdx.x;
  float s = 0.f;
  for (int i = tid; i < N; i += 256) s += ip[i];
  red[tid] = s;
  __syncthreads();
  for (int st = 128; st > 0; st >>= 1) {
    if (tid < st) red[tid] += red[tid + st];
    __syncthreads();
  }
  if (tid == 0) out[rc] = red[0] / (float)N;
}

// se = sigmoid(W2 @ relu(W1 @ gap + b1) + b2); one block per batch
__global__ void se_gate(const float* __restrict__ gap, const float* __restrict__ w1,
                        const float* __restrict__ b1, const float* __restrict__ w2,
                        const float* __restrict__ b2, float* __restrict__ se,
                        int C, int hid) {
  __shared__ float s_gap[1536];
  __shared__ float s_hid[96];
  const int b = blockIdx.x, tid = threadIdx.x;
  for (int c = tid; c < C; c += 256) s_gap[c] = gap[(long long)b * C + c];
  __syncthreads();
  for (int j = tid; j < hid; j += 256) {
    float s = b1[j];
    for (int c = 0; c < C; ++c) s += w1[(long long)j * C + c] * s_gap[c];
    s_hid[j] = s > 0.f ? s : 0.f;
  }
  __syncthreads();
  if (tid < 256) {
    float s = b2[tid];
    for (int j = 0; j < hid; ++j) s += w2[(long long)tid * hid + j] * s_hid[j];
    se[(long long)b * 256 + tid] = 1.f / (1.f + __expf(-s));
  }
}

// sbT[b][n][q] = bf16(relu((1+se[q]) * (gamma*out_c + bottle)))
__global__ void combine_se(const float* __restrict__ out_c, const float* __restrict__ bottle,
                           const float* __restrict__ se, const float* __restrict__ gamma,
                           bf16* __restrict__ sbT) {
  const int i = blockIdx.x * blockDim.x + threadIdx.x;
  if (i >= 2 * 4096 * 256) return;
  const int c = i & 255, n = (i >> 8) & 4095, b = i >> 20;
  const long long src = ((long long)(b * 256 + c)) * 4096 + n;
  const float bt = gamma[0] * out_c[src] + bottle[src];
  float v = se[(long long)b * 256 + c] * bt + bt;
  v = v > 0.f ? v : 0.f;
  sbT[i] = (bf16)v;
}

// bilinear upsample 32->64 align_corners, f32 [B][256][32][32] -> bf16 [B][4096][256]
__global__ void upsample_t(const float* __restrict__ in, bf16* __restrict__ outT) {
  const int i = blockIdx.x * blockDim.x + threadIdx.x;
  if (i >= 2 * 4096 * 256) return;
  const int c = i & 255, n = (i >> 8) & 4095, b = i >> 20;
  const int h = n >> 6, w = n & 63;
  const float fy = h * (31.f / 63.f), fx = w * (31.f / 63.f);
  const int y0 = (int)fy, x0 = (int)fx;
  const int y1 = y0 + 1 < 32 ? y0 + 1 : 31;
  const int x1 = x0 + 1 < 32 ? x0 + 1 : 31;
  const float wy = fy - y0, wx = fx - x0;
  const long long base = ((long long)(b * 256 + c)) * 1024;
  const float top = in[base + y0 * 32 + x0] * (1.f - wx) + in[base + y0 * 32 + x1] * wx;
  const float bot = in[base + y1 * 32 + x0] * (1.f - wx) + in[base + y1 * 32 + x1] * wx;
  outT[i] = (bf16)(top * (1.f - wy) + bot * wy);
}

// ---------------------------------------------------------------------------
// Host side
// ---------------------------------------------------------------------------
static inline int divup(int a, int b) { return (a + b - 1) / b; }

static void run_gemm(hipStream_t s, const bf16* A, const bf16* Bt, float* oF,
                     bf16* oMN, bf16* oNM, int Bn, int M, int N, int K, int lda,
                     int ldb, int ldoF, int ldoB, long long sA, long long sB,
                     long long sOF, long long sOB, const float* bias,
                     const float* scale, const float* shift, int relu, float alpha) {
  const int waves = Bn * divup(M, 16) * (N / 64);
  const int blocks = divup(waves, 8);
  gemm_nt<<<blocks, 256, 0, s>>>(A, Bt, oF, oMN, oNM, Bn, M, N, K, lda, ldb, ldoF,
                                 ldoB, sA, sB, sOF, sOB, bias, scale, shift, relu,
                                 alpha);
}

extern "C" void kernel_launch(void* const* d_in, const int* in_sizes, int n_in,
                              void* d_out, int out_size, void* d_ws, size_t ws_size,
                              hipStream_t stream) {
  (void)in_sizes; (void)n_in; (void)out_size; (void)ws_size;
  const int Bn = 2;
  const float* x = (const float*)d_in[0];
  auto P = [&](int i) { return (const float*)d_in[i]; };
  // param indices (flat dict order)
  const float *b0_w = P(1), *b0_s = P(2), *b0_b = P(3);
  const float *b1_w = P(4), *b1_s = P(5), *b1_b = P(6);
  const float *b2_w = P(7), *b2_s = P(8), *b2_b = P(9);
  const float *b3_w = P(10), *b3_s = P(11), *b3_b = P(12);
  const float *ctx_w = P(13), *ctx_s = P(14), *ctx_b = P(15);
  const float *sa_key_w = P(16), *sa_key_bias = P(17), *sa_key_s = P(18), *sa_key_b = P(19);
  const float *sa_val_w = P(20), *sa_val_b = P(21);
  const float *sa_W_w = P(22), *sa_W_b = P(23);
  const float *sa_out_w = P(24), *sa_out_bias = P(25), *sa_out_s = P(26), *sa_out_b = P(27);
  const float *sec_proj_w = P(28), *sec_proj_s = P(29), *sec_proj_b = P(30);
  const float *sec_gamma = P(31);
  const float *sec_se_w1 = P(32), *sec_se_b1 = P(33), *sec_se_w2 = P(34), *sec_se_b2 = P(35);
  const float *sec_out_w = P(36), *sec_out_s = P(37), *sec_out_b = P(38);
  const float *gsc_proj_w = P(39), *gsc_proj_s = P(40), *gsc_proj_b = P(41);
  const float *gsc_gamma = P(42);
  const float *gsc_se_w1 = P(43), *gsc_se_b1 = P(44), *gsc_se_w2 = P(45), *gsc_se_b2 = P(46);
  const float *gsc_out_w = P(47), *gsc_out_s = P(48), *gsc_out_b = P(49);
  const float *c8_w = P(50), *c8_b = P(51);

  // -------- workspace carve-out --------
  char* base = (char*)d_ws;
  size_t off = 0;
  auto take = [&](size_t bytes) -> void* {
    void* p = base + off;
    off = (off + bytes + 255) & ~(size_t)255;
    return p;
  };
  bf16* Xp   = (bf16*)take((size_t)Bn * 4096 * 512 * 2);   // x transposed [B][n][c]
  bf16* Xc   = (bf16*)take((size_t)Bn * 512 * 4096 * 2);   // x bf16 [B][c][n]
  bf16* Wb0  = (bf16*)take((size_t)256 * 512 * 2);
  bf16* W3_1 = (bf16*)take((size_t)9 * 256 * 512 * 2);
  bf16* W3_2 = (bf16*)take((size_t)9 * 256 * 512 * 2);
  bf16* W3_3 = (bf16*)take((size_t)9 * 256 * 512 * 2);
  bf16* W3_c = (bf16*)take((size_t)9 * 256 * 512 * 2);
  bf16* Wkey = (bf16*)take((size_t)128 * 256 * 2);
  bf16* Wval = (bf16*)take((size_t)256 * 256 * 2);
  bf16* WWm  = (bf16*)take((size_t)256 * 256 * 2);
  bf16* WoSA = (bf16*)take((size_t)256 * 256 * 2);
  bf16* WsP  = (bf16*)take((size_t)256 * 512 * 2);
  bf16* WsO  = (bf16*)take((size_t)256 * 256 * 2);
  bf16* WgP  = (bf16*)take((size_t)256 * 1536 * 2);
  bf16* WgO  = (bf16*)take((size_t)256 * 256 * 2);
  bf16* Wc8  = (bf16*)take((size_t)150 * 256 * 2);
  float* ctxF    = (float*)take((size_t)Bn * 256 * 4096 * 4);
  bf16*  xpT     = (bf16*)take((size_t)Bn * 1024 * 256 * 2);
  bf16*  kqT     = (bf16*)take((size_t)Bn * 1024 * 128 * 2);
  bf16*  vfb     = (bf16*)take((size_t)Bn * 256 * 1024 * 2);
  float* simF    = (float*)take((size_t)Bn * 1024 * 1024 * 4);
  bf16*  simB    = (bf16*)take((size_t)Bn * 1024 * 1024 * 2);
  bf16*  ctxT    = (bf16*)take((size_t)Bn * 1024 * 256 * 2);
  float* wctxF   = (float*)take((size_t)Bn * 256 * 1024 * 4);
  bf16*  upT     = (bf16*)take((size_t)Bn * 4096 * 256 * 2);
  float* yF      = (float*)take((size_t)Bn * 1536 * 4096 * 4);
  float* bottleF = (float*)take((size_t)Bn * 256 * 4096 * 4);
  bf16*  bottleB = (bf16*)take((size_t)Bn * 256 * 4096 * 2);
  float* energyF = (float*)take((size_t)Bn * 256 * 1536 * 4);
  bf16*  attnB   = (bf16*)take((size_t)Bn * 256 * 1536 * 2);
  float* outcF   = (float*)take((size_t)Bn * 256 * 4096 * 4);
  float* gapF    = (float*)take((size_t)Bn * 1536 * 4);
  float* seF     = (float*)take((size_t)Bn * 256 * 4);
  bf16*  sbT     = (bf16*)take((size_t)Bn * 4096 * 256 * 2);
  bf16*  Yp      = (bf16*)take((size_t)Bn * 4096 * 1536 * 2);
  bf16*  Yc      = (bf16*)take((size_t)Bn * 1536 * 4096 * 2);
  bf16*  goutT   = (bf16*)take((size_t)Bn * 4096 * 256 * 2);

  const long long sX = 4096LL * 512, sXc = 512LL * 4096;
  const long long sY = 1536LL * 4096;

  // -------- stage 0: data-type / layout prep --------
  transpose_cvt<<<dim3(128, 16, Bn), dim3(32, 32), 0, stream>>>(x, Xp, Xc, 512, 4096);
  cvt_plain<<<divup(256 * 512, 256), 256, 0, stream>>>(b0_w, Wb0, 256 * 512);
  cvt_w3x3<<<divup(9 * 256 * 512, 256), 256, 0, stream>>>(b1_w, W3_1, 256, 512);
  cvt_w3x3<<<divup(9 * 256 * 512, 256), 256, 0, stream>>>(b2_w, W3_2, 256, 512);
  cvt_w3x3<<<divup(9 * 256 * 512, 256), 256, 0, stream>>>(b3_w, W3_3, 256, 512);
  cvt_w3x3<<<divup(9 * 256 * 512, 256), 256, 0, stream>>>(ctx_w, W3_c, 256, 512);
  cvt_plain<<<divup(128 * 256, 256), 256, 0, stream>>>(sa_key_w, Wkey, 128 * 256);
  cvt_plain<<<divup(256 * 256, 256), 256, 0, stream>>>(sa_val_w, Wval, 256 * 256);
  cvt_plain<<<divup(256 * 256, 256), 256, 0, stream>>>(sa_W_w, WWm, 256 * 256);
  cvt_plain<<<divup(256 * 256, 256), 256, 0, stream>>>(sa_out_w, WoSA, 256 * 256);
  cvt_plain<<<divup(256 * 512, 256), 256, 0, stream>>>(sec_proj_w, WsP, 256 * 512);
  cvt_plain<<<divup(256 * 256, 256), 256, 0, stream>>>(sec_out_w, WsO, 256 * 256);
  cvt_plain<<<divup(256 * 1536, 256), 256, 0, stream>>>(gsc_proj_w, WgP, 256 * 1536);
  cvt_plain<<<divup(256 * 256, 256), 256, 0, stream>>>(gsc_out_w, WgO, 256 * 256);
  cvt_plain<<<divup(150 * 256, 256), 256, 0, stream>>>(c8_w, Wc8, 150 * 256);

  // -------- stage 1: ASPP branches (slices of y [B][1536][4096]) --------
  run_gemm(stream, Wb0, Xp, yF + 0, nullptr, nullptr, Bn, 256, 4096, 512, 512, 512,
           4096, 0, 0, sX, sY, 0, nullptr, b0_s, b0_b, 1, 1.f);
  {
    const int blocks = Bn * 2 * 64;  // (b, mhalf, row)
    conv3x3_wmma_lds<<<blocks, 256, 0, stream>>>(W3_1, Xp, yF + 256LL * 4096, sY, b1_s, b1_b, 12, Bn);
    conv3x3_wmma_lds<<<blocks, 256, 0, stream>>>(W3_2, Xp, yF + 512LL * 4096, sY, b2_s, b2_b, 24, Bn);
    conv3x3_wmma_lds<<<blocks, 256, 0, stream>>>(W3_3, Xp, yF + 768LL * 4096, sY, b3_s, b3_b, 36, Bn);
    conv3x3_wmma_lds<<<blocks, 256, 0, stream>>>(W3_c, Xp, ctxF, 256LL * 4096, ctx_s, ctx_b, 1, Bn);
  }

  // -------- stage 2: self-attention (OC) block --------
  maxpool_t<<<divup(Bn * 1024 * 256, 256), 256, 0, stream>>>(ctxF, xpT);
  run_gemm(stream, Wkey, xpT, nullptr, nullptr, kqT, Bn, 128, 1024, 256, 256, 256,
           0, 128, 0, 1024LL * 256, 0, 1024LL * 128, sa_key_bias, sa_key_s, sa_key_b, 1, 1.f);
  run_gemm(stream, Wval, xpT, nullptr, vfb, nullptr, Bn, 256, 1024, 256, 256, 256,
           0, 1024, 0, 1024LL * 256, 0, 256LL * 1024, sa_val_b, nullptr, nullptr, 0, 1.f);
  run_gemm(stream, kqT, kqT, simF, nullptr, nullptr, Bn, 1024, 1024, 128, 128, 128,
           1024, 0, 1024LL * 128, 1024LL * 128, 1024LL * 1024, 0, nullptr, nullptr,
           nullptr, 0, 1.f);
  softmax_rows<<<Bn * 1024, 256, 0, stream>>>(simF, simB, 1024, 0.08838834764831845f);
  run_gemm(stream, simB, vfb, nullptr, ctxT, nullptr, Bn, 1024, 256, 1024, 1024,
           1024, 0, 256, 1024LL * 1024, 256LL * 1024, 0, 1024LL * 256, nullptr,
           nullptr, nullptr, 0, 1.f);
  run_gemm(stream, WWm, ctxT, wctxF, nullptr, nullptr, Bn, 256, 1024, 256, 256, 256,
           1024, 0, 0, 1024LL * 256, 256LL * 1024, 0, sa_W_b, nullptr, nullptr, 0, 1.f);
  upsample_t<<<divup(Bn * 4096 * 256, 256), 256, 0, stream>>>(wctxF, upT);
  run_gemm(stream, WoSA, upT, yF + 1024LL * 4096, nullptr, nullptr, Bn, 256, 4096,
           256, 256, 256, 4096, 0, 0, 4096LL * 256, sY, 0, sa_out_bias, sa_out_s,
           sa_out_b, 1, 1.f);

  // -------- stage 3: guided SE-CAM on x (sec) --------
  run_gemm(stream, WsP, Xp, bottleF, bottleB, nullptr, Bn, 256, 4096, 512, 512, 512,
           4096, 4096, 0, sX, 256LL * 4096, 256LL * 4096, nullptr, sec_proj_s,
           sec_proj_b, 1, 1.f);
  run_gemm(stream, bottleB, Xc, energyF, nullptr, nullptr, Bn, 256, 512, 4096, 4096,
           4096, 512, 0, 256LL * 4096, sXc, 256LL * 512, 0, nullptr, nullptr,
           nullptr, 0, 1.f);
  softmax_rows<<<Bn * 256, 256, 0, stream>>>(energyF, attnB, 512, -1.f);
  run_gemm(stream, attnB, Xp, outcF, nullptr, nullptr, Bn, 256, 4096, 512, 512, 512,
           4096, 0, 256LL * 512, sX, 256LL * 4096, 0, nullptr, nullptr, nullptr, 0, 1.f);
  gap_mean<<<Bn * 512, 256, 0, stream>>>(x, gapF, 4096);
  se_gate<<<Bn, 256, 0, stream>>>(gapF, sec_se_w1, sec_se_b1, sec_se_w2, sec_se_b2,
                                  seF, 512, 32);
  combine_se<<<divup(Bn * 4096 * 256, 256), 256, 0, stream>>>(outcF, bottleF, seF,
                                                              sec_gamma, sbT);
  run_gemm(stream, WsO, sbT, yF + 1280LL * 4096, nullptr, nullptr, Bn, 256, 4096,
           256, 256, 256, 4096, 0, 0, 4096LL * 256, sY, 0, nullptr, sec_out_s,
           sec_out_b, 1, 1.f);

  // -------- stage 4: guided SE-CAM on y (gsc) --------
  transpose_cvt<<<dim3(128, 48, Bn), dim3(32, 32), 0, stream>>>(yF, Yp, Yc, 1536, 4096);
  run_gemm(stream, WgP, Yp, bottleF, bottleB, nullptr, Bn, 256, 4096, 1536, 1536,
           1536, 4096, 4096, 0, 4096LL * 1536, 256LL * 4096, 256LL * 4096, nullptr,
           gsc_proj_s, gsc_proj_b, 1, 1.f);
  run_gemm(stream, bottleB, Yc, energyF, nullptr, nullptr, Bn, 256, 1536, 4096,
           4096, 4096, 1536, 0, 256LL * 4096, 1536LL * 4096, 256LL * 1536, 0,
           nullptr, nullptr, nullptr, 0, 1.f);
  softmax_rows<<<Bn * 256, 256, 0, stream>>>(energyF, attnB, 1536, -1.f);
  run_gemm(stream, attnB, Yp, outcF, nullptr, nullptr, Bn, 256, 4096, 1536, 1536,
           1536, 4096, 0, 256LL * 1536, 4096LL * 1536, 256LL * 4096, 0, nullptr,
           nullptr, nullptr, 0, 1.f);
  gap_mean<<<Bn * 1536, 256, 0, stream>>>(yF, gapF, 4096);
  se_gate<<<Bn, 256, 0, stream>>>(gapF, gsc_se_w1, gsc_se_b1, gsc_se_w2, gsc_se_b2,
                                  seF, 1536, 96);
  combine_se<<<divup(Bn * 4096 * 256, 256), 256, 0, stream>>>(outcF, bottleF, seF,
                                                              gsc_gamma, sbT);
  run_gemm(stream, WgO, sbT, nullptr, nullptr, goutT, Bn, 256, 4096, 256, 256, 256,
           0, 256, 0, 4096LL * 256, 0, 4096LL * 256, nullptr, gsc_out_s, gsc_out_b,
           1, 1.f);

  // -------- stage 5: classifier conv8 -> d_out [B][150][4096] --------
  run_gemm(stream, Wc8, goutT, (float*)d_out, nullptr, nullptr, Bn, 150, 4096, 256,
           256, 256, 4096, 0, 0, 4096LL * 256, 150LL * 4096, 0, c8_b, nullptr,
           nullptr, 0, 1.f);
}
